// Policy_31104153158020
// MI455X (gfx1250) — matrix operand order repaired and checked by reading, not compiled, not run on hardware
//
#include <hip/hip_runtime.h>
#include <hip/hip_bf16.h>

typedef __attribute__((ext_vector_type(16))) _Float16 v16h;
typedef __attribute__((ext_vector_type(8)))  float    v8f;

// ---------------- constants ----------------
#define B_TOT   16384
#define M_TOK   200
#define NL      22
#define CPAD    32            // channel pad for WMMA K alignment
#define GRID    121           // 11*11
// workspace byte offsets (f16 repacked weights)
#define OFF_W1   0            // [128][25][32]  = 102400 f16
#define OFF_W2   204800       // [128][1152]    = 147456 f16
#define OFF_FC   499712       // [256][128]     = 32768  f16
#define OFF_SELF 565248       // [256][32]      = 8192   f16
#define OFF_HEAD 581632       // [32][512]      = 16384  f16
#define OFF_HB   614400       // [32] f32 head biases
#define PREP_N   307232       // total prep work items

__constant__ float g_maxv[NL] = {9.f,1.f,1.f,10.f,3.f,254.f,1.f,1.f,235.f,8.f,
                                 9.f,250.f,29.f,1.f,1.f,8.f,1.f,1.f,6.f,3.f,
                                 1.f,2.f};

union FragU { uint4 q[2]; v16h h; };

// A-fragment (16x32 f16): lane holds K = k0..k0+7 and k0+16..k0+23 (k0 = hi*8),
// caller passes p already offset by hi*8.
__device__ __forceinline__ v16h ld_a(const _Float16* p) {
    FragU u;
    u.q[0] = *(const uint4*)(p);
    u.q[1] = *(const uint4*)(p + 16);
    return u.h;
}
// B-fragment (32x16 f16): lane holds K = k0..k0+7 and k0+16..k0+23 (k0 = hi*8),
// caller passes p already offset by hi*8.
__device__ __forceinline__ v16h ld_b(const _Float16* p) {
    FragU u;
    u.q[0] = *(const uint4*)(p);
    u.q[1] = *(const uint4*)(p + 16);
    return u.h;
}

__device__ __forceinline__ v8f wmma32(v16h a, v16h b, v8f c) {
    return __builtin_amdgcn_wmma_f32_16x16x32_f16(false, a, false, b,
                                                  (short)0, c, false, false);
}

// ---------------- weight repack (f32 -> f16, fragment-friendly layouts) -----
__global__ void policy_prep(const float* __restrict__ c1w, const float* __restrict__ c2w,
                            const float* __restrict__ fcw, const float* __restrict__ sw,
                            const float* __restrict__ a0w, const float* __restrict__ a1w,
                            const float* __restrict__ vw,  const float* __restrict__ a0b,
                            const float* __restrict__ a1b, const float* __restrict__ vb,
                            char* __restrict__ ws)
{
    int idx = blockIdx.x * 256 + threadIdx.x;
    if (idx >= PREP_N) return;

    _Float16* W1h   = (_Float16*)(ws + OFF_W1);
    _Float16* W2h   = (_Float16*)(ws + OFF_W2);
    _Float16* FCh   = (_Float16*)(ws + OFF_FC);
    _Float16* SELFh = (_Float16*)(ws + OFF_SELF);
    _Float16* HEADh = (_Float16*)(ws + OFF_HEAD);
    float*    HBIAS = (float*)(ws + OFF_HB);

    if (idx < 102400) {                       // W1h[o][dh*5+dw][c]
        int o = idx / 800, rem = idx - o * 800;
        int pdw = rem >> 5, c = rem & 31;
        int dh = pdw / 5, dw = pdw - dh * 5;
        float v = (c < NL) ? c1w[((o * NL + c) * 5 + dh) * 5 + dw] : 0.f;
        W1h[idx] = (_Float16)v;
        return;
    }
    idx -= 102400;
    if (idx < 147456) {                       // W2h[o][c*9+p] == c2w linear
        W2h[idx] = (_Float16)c2w[idx];
        return;
    }
    idx -= 147456;
    if (idx < 32768) {                        // FCh[n][k]
        FCh[idx] = (_Float16)fcw[idx];
        return;
    }
    idx -= 32768;
    if (idx < 8192) {                         // SELFh[n][c(pad32)]
        int n = idx >> 5, c = idx & 31;
        SELFh[idx] = (_Float16)((c < NL) ? sw[n * NL + c] : 0.f);
        return;
    }
    idx -= 8192;
    if (idx < 16384) {                        // HEADh[n][k], rows: 0-8 a0, 9-18 a1, 19 v
        int n = idx >> 9, k = idx & 511;
        float v = 0.f;
        if (n < 9)       v = a0w[n * 512 + k];
        else if (n < 19) v = a1w[(n - 9) * 512 + k];
        else if (n == 19) v = vw[k];
        HEADh[idx] = (_Float16)v;
        return;
    }
    idx -= 16384;
    if (idx < 32) {
        float v = 0.f;
        if (idx < 9)       v = a0b[idx];
        else if (idx < 19) v = a1b[idx - 9];
        else if (idx == 19) v = vb[0];
        HBIAS[idx] = v;
    }
}

// ---------------- fused policy forward: 1 block = 16 batch rows -------------
__global__ __launch_bounds__(256, 1) void policy_main(
    const int* __restrict__ obs,
    const float* __restrict__ c1b, const float* __restrict__ c2b,
    const float* __restrict__ fcb, const float* __restrict__ selfb,
    const _Float16* __restrict__ W1h,  const _Float16* __restrict__ W2h,
    const _Float16* __restrict__ FCh,  const _Float16* __restrict__ SELFh,
    const _Float16* __restrict__ HEADh, const float* __restrict__ HBIAS,
    float* __restrict__ out)
{
    __shared__ __align__(16) _Float16 sFeat[16 * GRID * CPAD];  // [m][x][y][c]
    __shared__ __align__(16) _Float16 sSelf[16 * CPAD];         // center cell
    __shared__ __align__(16) _Float16 sH1[16 * 1152];           // [m][c*9+pos]
    __shared__ __align__(16) _Float16 sH2[16 * 128];            // [m][c]
    __shared__ __align__(16) _Float16 sHid[16 * 512];           // [m][self|cnn]

    const int tid  = threadIdx.x;
    const int lane = tid & 31;
    const int wv   = tid >> 5;        // 0..7 (wave32)
    const int hi   = lane >> 4;       // half-wave select
    const int ncol = lane & 15;       // WMMA column
    const int mrow = lane & 15;       // WMMA A-row
    const int b0   = blockIdx.x * 16;

    // ---- zero feat + self slice ----
    {
        uint4 z = make_uint4(0u, 0u, 0u, 0u);
        uint4* f4 = (uint4*)sFeat;
        for (int i = tid; i < (16 * GRID * CPAD) / 8; i += 256) f4[i] = z;
        uint4* s4 = (uint4*)sSelf;
        if (tid < (16 * CPAD) / 8) s4[tid] = z;
    }
    __syncthreads();

    // ---- token scatter (normalized, f16, channel-last) ----
    for (int t = tid; t < 16 * M_TOK; t += 256) {
        int m = t / M_TOK, tok = t - m * M_TOK;
        const int* o = obs + ((size_t)(b0 + m) * M_TOK + tok) * 3;
        int c0 = o[0], c1 = o[1], c2 = o[2];
        c0 = (c0 == 255) ? 0 : c0;
        c1 = (c1 == 255) ? 0 : c1;
        c2 = (c2 == 255) ? 0 : c2;
        int x = (c0 >> 4) & 15, y = c0 & 15;
        if (x < 11 && y < 11 && (unsigned)c1 < (unsigned)NL) {
            float v = (float)c2 / (g_maxv[c1] + 1e-8f);
            _Float16 hv = (_Float16)v;
            sFeat[(m * GRID + x * 11 + y) * CPAD + c1] = hv;
            if (x == 5 && y == 5) sSelf[m * CPAD + c1] = hv;
        }
    }
    __syncthreads();

    // ---- self encoder: Linear(22->256), K padded to 32 (1 WMMA / N-tile) ----
    {
        v16h a = ld_a(sSelf + mrow * CPAD + hi * 8);
        #pragma unroll
        for (int half = 0; half < 2; ++half) {
            int n = (wv * 2 + half) * 16 + ncol;            // 0..255
            float bias = selfb[n];
            v16h b = ld_b(SELFh + (size_t)n * 32 + hi * 8);
            v8f acc = {};
            acc = wmma32(a, b, acc);
            _Float16* dst = sHid + (8 * hi) * 512 + n;      // + r*512, const
            #pragma unroll
            for (int r = 0; r < 8; ++r)
                dst[r * 512] = (_Float16)fmaxf(acc[r] + bias, 0.f);
        }
    }

    // ---- conv1: split 9 positions into 3 row-groups of 3 (low VGPR) ----
    {
        const int cout = wv * 16 + ncol;                    // 0..127
        const float bias = c1b[cout];
        const _Float16* w1row   = W1h + (size_t)cout * 800 + hi * 8;
        const _Float16* featrow = sFeat + mrow * GRID * CPAD + hi * 8;
        #pragma unroll 1
        for (int ig = 0; ig < 3; ++ig) {                    // output row i
            v8f acc0 = {}, acc1 = {}, acc2 = {};
            for (int ks = 0; ks < 25; ++ks) {
                int dh = ks / 5, dw = ks - dh * 5;
                v16h b = ld_b(w1row + ks * 32);
                const _Float16* fr = featrow + ((ig * 3 + dh) * 11 + dw) * CPAD;
                acc0 = wmma32(ld_a(fr),            b, acc0);   // j=0 -> y=dw
                acc1 = wmma32(ld_a(fr + 3 * CPAD), b, acc1);   // j=1 -> y=3+dw
                acc2 = wmma32(ld_a(fr + 6 * CPAD), b, acc2);   // j=2 -> y=6+dw
            }
            _Float16* dst = sH1 + (8 * hi) * 1152 + cout * 9 + ig * 3; // + r*1152 + j
            #pragma unroll
            for (int r = 0; r < 8; ++r) {
                dst[r * 1152 + 0] = (_Float16)fmaxf(acc0[r] + bias, 0.f);
                dst[r * 1152 + 1] = (_Float16)fmaxf(acc1[r] + bias, 0.f);
                dst[r * 1152 + 2] = (_Float16)fmaxf(acc2[r] + bias, 0.f);
            }
        }
    }
    __syncthreads();

    // ---- conv2: K=1152 (36 K-steps), sH1 is already the im2col ----
    {
        const int cout = wv * 16 + ncol;
        const float bias = c2b[cout];
        const _Float16* wrow = W2h + (size_t)cout * 1152 + hi * 8;
        const _Float16* arow = sH1 + mrow * 1152 + hi * 8;
        v8f acc = {};
        #pragma unroll 4
        for (int ks = 0; ks < 36; ++ks)
            acc = wmma32(ld_a(arow + ks * 32), ld_b(wrow + ks * 32), acc);
        _Float16* dst = sH2 + (8 * hi) * 128 + cout;        // + r*128
        #pragma unroll
        for (int r = 0; r < 8; ++r)
            dst[r * 128] = (_Float16)fmaxf(acc[r] + bias, 0.f);
    }
    __syncthreads();

    // ---- fc: Linear(128->256), K=4 steps, wave does 2 N-tiles ----
    {
        const _Float16* arow = sH2 + mrow * 128 + hi * 8;
        #pragma unroll 1
        for (int half = 0; half < 2; ++half) {
            int n = (wv * 2 + half) * 16 + ncol;            // 0..255
            float bias = fcb[n];
            const _Float16* wrow = FCh + (size_t)n * 128 + hi * 8;
            v8f acc = {};
            #pragma unroll
            for (int ks = 0; ks < 4; ++ks)
                acc = wmma32(ld_a(arow + ks * 32), ld_b(wrow + ks * 32), acc);
            _Float16* dst = sHid + (8 * hi) * 512 + 256 + n; // + r*512
            #pragma unroll
            for (int r = 0; r < 8; ++r)
                dst[r * 512] = (_Float16)fmaxf(acc[r] + bias, 0.f);
        }
    }
    __syncthreads();

    // ---- heads: [16x512] x [512x32] (20 valid cols), waves 0-1 ----
    if (wv < 2) {
        int n = wv * 16 + ncol;                             // 0..31
        float bias = HBIAS[n];
        const _Float16* wrow = HEADh + (size_t)n * 512 + hi * 8;
        const _Float16* arow = sHid + mrow * 512 + hi * 8;
        v8f acc = {};
        #pragma unroll
        for (int ks = 0; ks < 16; ++ks)
            acc = wmma32(ld_a(arow + ks * 32), ld_b(wrow + ks * 32), acc);
        #pragma unroll
        for (int r = 0; r < 8; ++r) {
            int bidx = b0 + r + 8 * hi;
            float v = acc[r] + bias;
            if (n < 9)        out[(size_t)bidx * 9 + n] = v;                 // a0
            else if (n < 19)  out[147456 + (size_t)bidx * 10 + (n - 9)] = v; // a1
            else if (n == 19) out[311296 + bidx] = v;                        // value
        }
    }
}

extern "C" void kernel_launch(void* const* d_in, const int* in_sizes, int n_in,
                              void* d_out, int out_size, void* d_ws, size_t ws_size,
                              hipStream_t stream) {
    const int*   obs = (const int*)d_in[0];
    const float* c1w = (const float*)d_in[1];
    const float* c1b = (const float*)d_in[2];
    const float* c2w = (const float*)d_in[3];
    const float* c2b = (const float*)d_in[4];
    const float* fcw = (const float*)d_in[5];
    const float* fcb = (const float*)d_in[6];
    const float* sw  = (const float*)d_in[7];
    const float* sb  = (const float*)d_in[8];
    const float* a0w = (const float*)d_in[9];
    const float* a0b = (const float*)d_in[10];
    const float* a1w = (const float*)d_in[11];
    const float* a1b = (const float*)d_in[12];
    const float* vw  = (const float*)d_in[13];
    const float* vb  = (const float*)d_in[14];

    char* ws = (char*)d_ws;
    float* out = (float*)d_out;

    policy_prep<<<(PREP_N + 255) / 256, 256, 0, stream>>>(
        c1w, c2w, fcw, sw, a0w, a1w, vw, a0b, a1b, vb, ws);

    const _Float16* W1h   = (const _Float16*)(ws + OFF_W1);
    const _Float16* W2h   = (const _Float16*)(ws + OFF_W2);
    const _Float16* FCh   = (const _Float16*)(ws + OFF_FC);
    const _Float16* SELFh = (const _Float16*)(ws + OFF_SELF);
    const _Float16* HEADh = (const _Float16*)(ws + OFF_HEAD);
    const float*    HBIAS = (const float*)(ws + OFF_HB);

    policy_main<<<B_TOT / 16, 256, 0, stream>>>(
        obs, c1b, c2b, fcb, sb, W1h, W2h, FCh, SELFh, HEADh, HBIAS, out);
}
